// _FastSelfAttention_5540507812364
// MI455X (gfx1250) — matrix operand order repaired and checked by reading, compiled, not run on hardware
//
#include <hip/hip_runtime.h>
#include <hip/hip_bf16.h>

#define B_  8
#define S_  4096
#define D_  1024
#define H_  16
#define DH_ 64
#define M_  (B_ * S_)

typedef __attribute__((ext_vector_type(16))) __bf16 bf16x16;
typedef __attribute__((ext_vector_type(8)))  float  f32x8;

union FragBF { uint4 u[2]; bf16x16 v; };

__device__ __forceinline__ unsigned short f2bf(float f) {
  union { float f; unsigned int u; } x; x.f = f;
  unsigned int r = x.u + 0x7FFFu + ((x.u >> 16) & 1u);   // round-to-nearest-even
  return (unsigned short)(r >> 16);
}

__device__ __forceinline__ f32x8 zero8() {
  f32x8 z;
  #pragma unroll
  for (int i = 0; i < 8; ++i) z[i] = 0.0f;
  return z;
}

// ---------------------------------------------------------------------------
// One-time weight conversion: W[K][NN] f32  ->  WT[NN][K] bf16 (transposed).
// ---------------------------------------------------------------------------
template<int NN>
__global__ __launch_bounds__(256) void convert_wt(
    const float* __restrict__ W, unsigned short* __restrict__ WT, int K)
{
  const unsigned idx = blockIdx.x * 256u + threadIdx.x;   // over K*NN
  const unsigned n = idx & (NN - 1);
  const unsigned k = idx >> (NN == 1024 ? 10 : 4);
  WT[n * (unsigned)K + k] = f2bf(W[idx]);
}

// ---------------------------------------------------------------------------
// C[M,N] = (A * scaleVec?) @ W + bias (+ addc).  A f32 -> bf16 in LDS,
// W pre-converted bf16 [N][K].  256 thr / 8 waves, 128x128 tile, K-tile 32.
// Double-buffered LDS; only aReg (16 VGPRs) lives across the 8 WMMAs:
//   barrier -> loadB(i+1), loadA(i+1), storeB(next)   [B wait is partial]
//           -> 8 WMMAs on buf[cur] -> scale+convert+storeA(next)
// ---------------------------------------------------------------------------
template<bool HAS_SCALE, bool HAS_ADD>
__global__ __launch_bounds__(256, 1) void gemm128(
    const float* __restrict__ A, const unsigned short* __restrict__ WT,
    const float* __restrict__ bias, float* __restrict__ C,
    const float* __restrict__ sv, const float* __restrict__ addc,
    int M, int N, int K, int rowsPerBatch)
{
  constexpr int PAD = 40;                      // 80B row stride: 16 banks, 16B aligned
  __shared__ __align__(16) unsigned short ldsA[2][128 * PAD];
  __shared__ __align__(16) unsigned short ldsB[2][128 * PAD];

  const int tid    = threadIdx.x;
  const int blockN = blockIdx.x * 128;
  const int blockM = blockIdx.y * 128;
  const int lane   = tid & 31;
  const int wave   = tid >> 5;
  const int waveM  = (wave & 1) * 64;
  const int waveN  = (wave >> 1) * 32;
  const int half   = lane >> 4;
  const int l15    = lane & 15;

  f32x8 acc[4][2];
  #pragma unroll
  for (int i = 0; i < 4; ++i)
    #pragma unroll
    for (int j = 0; j < 2; ++j) acc[i][j] = zero8();

  // staging map: thread -> (row = tid>>1, 16-wide k chunk = (tid&1)*16)
  const int rowA  = tid >> 1;
  const int kOff  = (tid & 1) << 4;
  const float* aRow = A + (unsigned)(blockM + rowA) * (unsigned)K + kOff;
  const float* svRow = HAS_SCALE
      ? sv + (unsigned)((blockM + rowA) / rowsPerBatch) * (unsigned)K + kOff
      : nullptr;
  const unsigned short* wRow = WT + (unsigned)(blockN + rowA) * (unsigned)K + kOff;

  float4 aReg[4];
  uint4  bReg[2];

  auto loadB = [&](int kb) {                 // issue FIRST (short L2 latency)
    #pragma unroll
    for (int j = 0; j < 2; ++j) bReg[j] = *(const uint4*)(wRow + kb + 8 * j);
  };
  auto loadA = [&](int kb) {                 // HBM stream; stays in flight thru WMMAs
    #pragma unroll
    for (int j = 0; j < 4; ++j) aReg[j] = *(const float4*)(aRow + kb + 4 * j);
  };
  auto storeB = [&](int buf) {               // waits only on B loads (in-order cnt)
    unsigned short* dB = &ldsB[buf][rowA * PAD + kOff];
    #pragma unroll
    for (int j = 0; j < 2; ++j) *(uint4*)(dB + 8 * j) = bReg[j];
  };
  auto storeA = [&](int buf, int kb) {       // after compute: scale+convert+store
    unsigned short* dA = &ldsA[buf][rowA * PAD + kOff];
    #pragma unroll
    for (int j = 0; j < 4; ++j) {
      float4 v = aReg[j];
      if (HAS_SCALE) {
        const float4 s4 = *(const float4*)(svRow + kb + 4 * j);  // L0-hot 4KB row
        v.x *= s4.x; v.y *= s4.y; v.z *= s4.z; v.w *= s4.w;
      }
      ushort4 p;
      p.x = f2bf(v.x); p.y = f2bf(v.y); p.z = f2bf(v.z); p.w = f2bf(v.w);
      *(ushort4*)(dA + 4 * j) = p;
    }
  };

  loadB(0); loadA(0);
  storeB(0); storeA(0, 0);
  int cur = 0;

  for (int kb = 0; kb < K; kb += 32) {
    __syncthreads();                       // buf[cur] ready; buf[cur^1] free
    const bool more = (kb + 32) < K;
    if (more) {
      loadB(kb + 32);
      loadA(kb + 32);
      storeB(cur ^ 1);                     // partial loadcnt wait (A still in flight)
    }

    FragBF bF0, bF1;
    {
      const unsigned short* p0 = &ldsB[cur][(waveN + l15) * PAD + half * 16];
      bF0.u[0] = *(const uint4*)p0;
      bF0.u[1] = *(const uint4*)(p0 + 8);
      const unsigned short* p1 = &ldsB[cur][(waveN + 16 + l15) * PAD + half * 16];
      bF1.u[0] = *(const uint4*)p1;
      bF1.u[1] = *(const uint4*)(p1 + 8);
    }
    #pragma unroll
    for (int mt = 0; mt < 4; ++mt) {
      FragBF aF;
      const unsigned short* p = &ldsA[cur][(waveM + mt * 16 + l15) * PAD + half * 16];
      aF.u[0] = *(const uint4*)p;
      aF.u[1] = *(const uint4*)(p + 8);
      acc[mt][0] = __builtin_amdgcn_wmma_f32_16x16x32_bf16(
          false, aF.v, false, bF0.v, (short)0, acc[mt][0], false, false);
      acc[mt][1] = __builtin_amdgcn_wmma_f32_16x16x32_bf16(
          false, aF.v, false, bF1.v, (short)0, acc[mt][1], false, false);
    }

    if (more) storeA(cur ^ 1, kb + 32);    // A loadcnt wait lands here, after WMMAs
    cur ^= 1;
  }

  // epilogue (32-bit indices)
  #pragma unroll
  for (int nt = 0; nt < 2; ++nt) {
    const unsigned col = (unsigned)(blockN + waveN + nt * 16 + l15);
    const float bv = bias[col];
    #pragma unroll
    for (int mt = 0; mt < 4; ++mt) {
      const unsigned row0 = (unsigned)(blockM + waveM + mt * 16 + half * 8);
      unsigned idx = row0 * (unsigned)N + col;
      #pragma unroll
      for (int r = 0; r < 8; ++r) {
        float vv = acc[mt][nt][r] + bv;
        if (HAS_ADD) vv += addc[idx];
        C[idx] = vv;
        idx += (unsigned)N;
      }
    }
  }
}

// ---------------------------------------------------------------------------
// Score GEMM: out(B,H,S) = transpose(((A*sv?) @ Wp + bias) * scale) + mask
// Wp pre-converted bf16 [16][K] cached whole in LDS.  128 thr / 4 waves.
// ---------------------------------------------------------------------------
template<bool HAS_SCALE>
__global__ __launch_bounds__(128, 1) void score16(
    const float* __restrict__ A, const unsigned short* __restrict__ WTp,
    const float* __restrict__ biasp, const float* __restrict__ mask,
    float* __restrict__ out, const float* __restrict__ sv,
    int K, float scale)
{
  __shared__ __align__(16) unsigned short ldsW[16 * 1032];  // [n][k], pad 1024->1032
  __shared__ __align__(16) unsigned short ldsA[2][64 * 40];

  const int tid  = threadIdx.x;
  const int lane = tid & 31;
  const int w    = tid >> 5;
  const int half = lane >> 4;
  const int l15  = lane & 15;
  const int m0   = blockIdx.x * 64;

  // whole weight (16 x 1024 bf16) into padded LDS via b128 moves
  for (int i = tid; i < 2048; i += 128) {            // 2048 uint4 = 16K bf16
    const int n = i >> 7, j = i & 127;
    *(uint4*)&ldsW[n * 1032 + j * 8] = *(const uint4*)(WTp + (unsigned)n * (unsigned)K + j * 8);
  }

  f32x8 acc = zero8();
  const int rowA = tid >> 1;
  const int kOff = (tid & 1) << 4;
  const float* aRow  = A + (unsigned)(m0 + rowA) * (unsigned)K + kOff;
  const float* svRow = HAS_SCALE
      ? sv + (unsigned)((m0 + rowA) / S_) * (unsigned)K + kOff : nullptr;

  float4 aReg[4];
  auto loadA = [&](int kb) {
    #pragma unroll
    for (int j = 0; j < 4; ++j) aReg[j] = *(const float4*)(aRow + kb + 4 * j);
  };
  auto storeA = [&](int buf, int kb) {
    unsigned short* dst = &ldsA[buf][rowA * 40 + kOff];
    #pragma unroll
    for (int j = 0; j < 4; ++j) {
      float4 v = aReg[j];
      if (HAS_SCALE) {
        const float4 s4 = *(const float4*)(svRow + kb + 4 * j);
        v.x *= s4.x; v.y *= s4.y; v.z *= s4.z; v.w *= s4.w;
      }
      ushort4 p;
      p.x = f2bf(v.x); p.y = f2bf(v.y); p.z = f2bf(v.z); p.w = f2bf(v.w);
      *(ushort4*)(dst + 4 * j) = p;
    }
  };

  loadA(0);
  storeA(0, 0);
  int cur = 0;

  for (int kb = 0; kb < K; kb += 32) {
    __syncthreads();
    const bool more = (kb + 32) < K;
    if (more) loadA(kb + 32);

    FragBF aF, bF;
    const unsigned short* pa = &ldsA[cur][(w * 16 + l15) * 40 + half * 16];
    aF.u[0] = *(const uint4*)pa;
    aF.u[1] = *(const uint4*)(pa + 8);
    const unsigned short* pb = &ldsW[l15 * 1032 + kb + half * 16];
    bF.u[0] = *(const uint4*)pb;
    bF.u[1] = *(const uint4*)(pb + 8);
    acc = __builtin_amdgcn_wmma_f32_16x16x32_bf16(
        false, aF.v, false, bF.v, (short)0, acc, false, false);

    if (more) storeA(cur ^ 1, kb + 32);
    cur ^= 1;
  }

  const int   h  = l15;
  const float bb = biasp[h];
  const unsigned b = (unsigned)m0 / S_;
  #pragma unroll
  for (int r = 0; r < 8; ++r) {
    const unsigned m = (unsigned)(m0 + w * 16 + half * 8 + r);
    const unsigned s = m & (S_ - 1);
    out[(b * H_ + h) * S_ + s] = (acc[r] + bb) * scale + mask[b * S_ + s];
  }
}

// ---------------------------------------------------------------------------
// Fused softmax-over-S + head pooling (float4-vectorized data loop):
// pooled[b, h*DH + d] = (sum_s softmax(score[b,h,:])[s] * data[b,s,h*DH+d]) * sv?
// ---------------------------------------------------------------------------
template<bool HAS_SCALE>
__global__ __launch_bounds__(256) void pool_softmax(
    const float* __restrict__ score,   // (B,H,S)
    const float* __restrict__ data,    // (B,S,D)
    const float* __restrict__ sv,      // (B,D) or null
    float* __restrict__ pooled)        // (B,D)
{
  __shared__ float  lw[S_];
  __shared__ float  red[256];
  __shared__ float4 red4[256];
  const int tid = threadIdx.x;
  const unsigned bh = blockIdx.x;
  const unsigned b  = bh / H_, h = bh % H_;
  const float* srow = score + bh * S_;

  float mx = -3.0e38f;
  for (int s = tid; s < S_; s += 256) mx = fmaxf(mx, srow[s]);
  red[tid] = mx; __syncthreads();
  for (int off = 128; off > 0; off >>= 1) {
    if (tid < off) red[tid] = fmaxf(red[tid], red[tid + off]);
    __syncthreads();
  }
  mx = red[0]; __syncthreads();

  float sum = 0.0f;
  for (int s = tid; s < S_; s += 256) {
    const float e = __expf(srow[s] - mx);
    lw[s] = e; sum += e;
  }
  red[tid] = sum; __syncthreads();
  for (int off = 128; off > 0; off >>= 1) {
    if (tid < off) red[tid] += red[tid + off];
    __syncthreads();
  }
  const float inv = 1.0f / red[0];
  __syncthreads();

  // weighted sum: 16 column-quads x 16 s-groups, float4 loads
  const int colq = tid & 15;             // which 4-column group
  const int grp  = tid >> 4;             // s stride group (16)
  const float* dptr = data + b * (S_ * D_) + h * DH_ + colq * 4;
  float4 part; part.x = part.y = part.z = part.w = 0.0f;
  for (int s = grp; s < S_; s += 16) {
    const float  wgt = lw[s];
    const float4 v = *(const float4*)(dptr + (unsigned)s * D_);
    part.x += wgt * v.x; part.y += wgt * v.y;
    part.z += wgt * v.z; part.w += wgt * v.w;
  }
  red4[tid] = part; __syncthreads();
  for (int off = 8; off > 0; off >>= 1) {
    if (tid < off * 16) {
      float4 o = red4[tid + off * 16];
      red4[tid].x += o.x; red4[tid].y += o.y;
      red4[tid].z += o.z; red4[tid].w += o.w;
    }
    __syncthreads();
  }
  if (tid < 16) {
    float4 tot = red4[tid];
    tot.x *= inv; tot.y *= inv; tot.z *= inv; tot.w *= inv;
    const unsigned o = b * D_ + h * DH_ + tid * 4;
    if (HAS_SCALE) {
      const float4 s4 = *(const float4*)(sv + o);
      tot.x *= s4.x; tot.y *= s4.y; tot.z *= s4.z; tot.w *= s4.w;
    }
    *(float4*)(pooled + o) = tot;
  }
}

// ---------------------------------------------------------------------------
extern "C" void kernel_launch(void* const* d_in, const int* in_sizes, int n_in,
                              void* d_out, int out_size, void* d_ws, size_t ws_size,
                              hipStream_t stream)
{
  (void)in_sizes; (void)n_in; (void)out_size; (void)ws_size;
  const float* hs   = (const float*)d_in[0];
  const float* mask = (const float*)d_in[1];
  const float* Wq   = (const float*)d_in[2];
  const float* bq   = (const float*)d_in[3];
  const float* Wqa  = (const float*)d_in[4];
  const float* bqa  = (const float*)d_in[5];
  const float* Wk   = (const float*)d_in[6];
  const float* bk   = (const float*)d_in[7];
  const float* Wka  = (const float*)d_in[8];
  const float* bka  = (const float*)d_in[9];
  const float* Wt   = (const float*)d_in[10];
  const float* bt   = (const float*)d_in[11];
  float* out = (float*)d_out;

  float* ws = (float*)d_ws;
  float* q  = ws;                           // M*D f32
  float* k  = q  + (size_t)M_ * D_;         // M*D f32
  float* qs = k  + (size_t)M_ * D_;         // B*H*S f32
  float* ks = qs + (size_t)B_ * H_ * S_;    // B*H*S f32
  float* pq = ks + (size_t)B_ * H_ * S_;    // B*D f32
  float* pk = pq + (size_t)B_ * D_;         // B*D f32
  unsigned short* wtq  = (unsigned short*)(pk + (size_t)B_ * D_);  // D*D bf16
  unsigned short* wtk  = wtq  + (size_t)D_ * D_;
  unsigned short* wtt  = wtk  + (size_t)D_ * D_;
  unsigned short* wtqa = wtt  + (size_t)D_ * D_;                   // H*D bf16
  unsigned short* wtka = wtqa + (size_t)H_ * D_;

  const float scale = 0.125f;               // 1/sqrt(DH=64)

  // one-time weight transposes/conversions (f32 [K][N] -> bf16 [N][K])
  convert_wt<D_><<<dim3(D_ * D_ / 256), dim3(256), 0, stream>>>(Wq,  wtq,  D_);
  convert_wt<D_><<<dim3(D_ * D_ / 256), dim3(256), 0, stream>>>(Wk,  wtk,  D_);
  convert_wt<D_><<<dim3(D_ * D_ / 256), dim3(256), 0, stream>>>(Wt,  wtt,  D_);
  convert_wt<H_><<<dim3(D_ * H_ / 256), dim3(256), 0, stream>>>(Wqa, wtqa, D_);
  convert_wt<H_><<<dim3(D_ * H_ / 256), dim3(256), 0, stream>>>(Wka, wtka, D_);

  const dim3 blkG(256), grdG(D_ / 128, M_ / 128);
  const dim3 blkS(128), grdS(M_ / 64);
  const dim3 blkP(256), grdP(B_ * H_);

  // q = hs @ Wq + bq ; k = hs @ Wk + bk
  gemm128<false, false><<<grdG, blkG, 0, stream>>>(hs, wtq, bq, q, nullptr, nullptr, M_, D_, D_, S_);
  gemm128<false, false><<<grdG, blkG, 0, stream>>>(hs, wtk, bk, k, nullptr, nullptr, M_, D_, D_, S_);

  // q_score -> softmax+pool -> pooled_q (B,D)
  score16<false><<<grdS, blkS, 0, stream>>>(q, wtqa, bqa, mask, qs, nullptr, D_, scale);
  pool_softmax<false><<<grdP, blkP, 0, stream>>>(qs, q, nullptr, pq);

  // mixed_qk = k * pq (fused scale vector): k_score -> softmax+pool -> pooled_k
  score16<true><<<grdS, blkS, 0, stream>>>(k, wtka, bka, mask, ks, pq, D_, scale);
  pool_softmax<true><<<grdP, blkP, 0, stream>>>(ks, k, pq, pk);

  // out = (q * pk) @ Wt + bt + q   (weighted never materialized)
  gemm128<true, true><<<grdG, blkG, 0, stream>>>(q, wtt, bt, out, pk, q, M_, D_, D_, S_);
}
